// GCN3_83227876262526
// MI455X (gfx1250) — compile-verified, matrix operand order checked by reading, not run on hardware
//
#include <hip/hip_runtime.h>

#define NN 50000
#define NE 400000
#define BN_EPS 1e-5f
#define NEG_SLOPE 0.1f

typedef __attribute__((ext_vector_type(16))) __bf16 v16bf;
typedef __attribute__((ext_vector_type(8)))  float v8f;

__device__ __forceinline__ unsigned bfbits(float f) {
  unsigned u = __builtin_bit_cast(unsigned, f);
  return (u + 0x7FFFu + ((u >> 16) & 1u)) >> 16;     // RNE f32 -> bf16 bits
}
__device__ __forceinline__ __bf16 f2bf(float f) {
  unsigned short h = (unsigned short)bfbits(f);
  return __builtin_bit_cast(__bf16, h);
}

// ---------------- elementwise helpers ----------------
__global__ void k_zero(float* __restrict__ p, long long n) {
  long long i = (long long)blockIdx.x * blockDim.x + threadIdx.x;
  long long stride = (long long)gridDim.x * blockDim.x;
  for (; i < n; i += stride) p[i] = 0.0f;
}

__global__ void k_count(const int* __restrict__ dst, float* __restrict__ cnt, int e) {
  int i = blockIdx.x * blockDim.x + threadIdx.x;
  if (i < e) atomicAdd(&cnt[dst[i]], 1.0f);
}

__global__ void k_invert(float* __restrict__ inv, int n) {
  int i = blockIdx.x * blockDim.x + threadIdx.x;
  if (i < n) inv[i] = 1.0f / fmaxf(inv[i], 1.0f);
}

// scatter-add: agg[dst[e], f] += x[src[e], f]
__global__ void k_scatter_add(const float* __restrict__ x, const int* __restrict__ src,
                              const int* __restrict__ dst, float* __restrict__ agg,
                              int e, int din) {
  long long n = (long long)e * din;
  long long i = (long long)blockIdx.x * blockDim.x + threadIdx.x;
  long long stride = (long long)gridDim.x * blockDim.x;
  for (; i < n; i += stride) {
    int ed = (int)(i / din);
    int f  = (int)(i % din);
    atomicAdd(&agg[(long long)dst[ed] * din + f], x[(long long)src[ed] * din + f]);
  }
}

// x (f32) -> bf16
__global__ void k_to_bf16(const float* __restrict__ in, __bf16* __restrict__ out,
                          long long n) {
  long long i = (long long)blockIdx.x * blockDim.x + threadIdx.x;
  long long stride = (long long)gridDim.x * blockDim.x;
  for (; i < n; i += stride) out[i] = f2bf(in[i]);
}

// agg (f32) * inv[node] -> bf16  (fuses the mean normalization)
__global__ void k_agg_to_bf16(const float* __restrict__ agg, const float* __restrict__ inv,
                              __bf16* __restrict__ out, int n, int din) {
  long long tot = (long long)n * din;
  long long i = (long long)blockIdx.x * blockDim.x + threadIdx.x;
  long long stride = (long long)gridDim.x * blockDim.x;
  for (; i < tot; i += stride) {
    int node = (int)(i / din);
    out[i] = f2bf(agg[i] * inv[node]);
  }
}

// transpose + convert weights: Wt[n*din + k] = bf16(W[k*dout + n])
__global__ void k_w_transpose(const float* __restrict__ W, __bf16* __restrict__ Wt,
                              int din, int dout) {
  int tot = din * dout;
  int i = blockIdx.x * blockDim.x + threadIdx.x;
  if (i < tot) {
    int n = i / din, k = i % din;
    Wt[i] = f2bf(W[(long long)k * dout + n]);
  }
}

// ---------------- fused SAGE GEMM (din % 32 == 0, dout % 64 == 0) ----------------
// out = act(Abf@Wl + Xbf@Wr + bl); all operands pre-converted bf16, weights transposed.
// Each wave owns 16 rows x 64 cols (4 WMMA tiles): 1 A fragment feeds 4 WMMAs per chunk.
// grid.x = ceil(NN/64) (4 waves = 4 row tiles), grid.y = dout/64. Block = 128 threads.
__global__ void k_sage_gemm(const __bf16* __restrict__ Abf, const __bf16* __restrict__ Xbf,
                            const __bf16* __restrict__ Wlt, const float* __restrict__ bl,
                            const __bf16* __restrict__ Wrt,
                            float* __restrict__ out, int din, int dout, int act) {
  int wave = threadIdx.x >> 5;
  int lane = threadIdx.x & 31;
  int m0 = blockIdx.x * 64 + wave * 16;
  if (m0 >= NN) return;                   // wave-uniform
  int n0 = blockIdx.y * 64;
  int lrow = lane & 15;
  int hi   = lane >> 4;
  int row  = m0 + lrow;

  v8f acc[4] = {v8f{}, v8f{}, v8f{}, v8f{}};
#pragma unroll 1
  for (int pass = 0; pass < 2; ++pass) {
    const __bf16* A  = pass ? Xbf : Abf;
    const __bf16* Wt = pass ? Wrt : Wlt;
    const __bf16* arow = A  + (long long)row * din;
    const __bf16* w0   = Wt + (long long)(n0 + lrow) * din;  // tile j at +16*j*din
    for (int k0 = 0; k0 < din; k0 += 32) {
      // fragment K pattern: k0 + 8*hi + {0..7}, k0 + 16 + 8*hi + {0..7} (contiguous bf16)
      const __bf16* ap = arow + k0 + (hi << 3);
      union { v16bf v; uint4 q[2]; } af, b0, b1, b2, b3;
      af.q[0] = *(const uint4*)(ap);
      af.q[1] = *(const uint4*)(ap + 16);
      const __bf16* wp = w0 + k0 + (hi << 3);
      b0.q[0] = *(const uint4*)(wp);
      b0.q[1] = *(const uint4*)(wp + 16);
      b1.q[0] = *(const uint4*)(wp + 16 * din);
      b1.q[1] = *(const uint4*)(wp + 16 * din + 16);
      b2.q[0] = *(const uint4*)(wp + 32 * din);
      b2.q[1] = *(const uint4*)(wp + 32 * din + 16);
      b3.q[0] = *(const uint4*)(wp + 48 * din);
      b3.q[1] = *(const uint4*)(wp + 48 * din + 16);
      acc[0] = __builtin_amdgcn_wmma_f32_16x16x32_bf16(false, af.v, false, b0.v,
                                                       (short)0, acc[0], false, false);
      acc[1] = __builtin_amdgcn_wmma_f32_16x16x32_bf16(false, af.v, false, b1.v,
                                                       (short)0, acc[1], false, false);
      acc[2] = __builtin_amdgcn_wmma_f32_16x16x32_bf16(false, af.v, false, b2.v,
                                                       (short)0, acc[2], false, false);
      acc[3] = __builtin_amdgcn_wmma_f32_16x16x32_bf16(false, af.v, false, b3.v,
                                                       (short)0, acc[3], false, false);
    }
  }

#pragma unroll
  for (int j = 0; j < 4; ++j) {
    int col = n0 + 16 * j + lrow;
    float b = bl[col];
#pragma unroll
    for (int r = 0; r < 8; ++r) {
      int mrow = m0 + r + (hi << 3);      // C/D layout: vgpr r -> M = r + 8*hi
      float v = acc[j][r] + b;
      if (act == 1) v = v > 0.0f ? v : 0.0f;
      out[(long long)mrow * dout + col] = v;
    }
  }
}

// layer 0 (din=3, dout=64): tiny GEMM, plain VALU, fused relu
__global__ void k_sage3(const float* __restrict__ agg3, const float* __restrict__ x,
                        const float* __restrict__ inv,
                        const float* __restrict__ Wl, const float* __restrict__ bl,
                        const float* __restrict__ Wr, float* __restrict__ out, int n) {
  int i = blockIdx.x * blockDim.x + threadIdx.x;
  if (i >= n * 64) return;
  int node = i >> 6, c = i & 63;
  float s = inv[node];
  float v = bl[c];
#pragma unroll
  for (int k = 0; k < 3; ++k)
    v += agg3[node * 3 + k] * s * Wl[k * 64 + c] + x[node * 3 + k] * Wr[k * 64 + c];
  out[(long long)node * 64 + c] = fmaxf(v, 0.0f);
}

// ---------------- batchnorm (training stats) ----------------
__global__ void k_bn_stats(const float* __restrict__ x, float* __restrict__ mean,
                           float* __restrict__ var, int n, int C) {
  int c = blockIdx.x;
  float s = 0.0f, s2 = 0.0f;
  for (int i = threadIdx.x; i < n; i += blockDim.x) {
    float v = x[(long long)i * C + c];
    s += v; s2 += v * v;
  }
  __shared__ float sh[256], sh2[256];
  sh[threadIdx.x] = s; sh2[threadIdx.x] = s2;
  __syncthreads();
  for (int o = blockDim.x >> 1; o > 0; o >>= 1) {
    if (threadIdx.x < (unsigned)o) {
      sh[threadIdx.x]  += sh[threadIdx.x + o];
      sh2[threadIdx.x] += sh2[threadIdx.x + o];
    }
    __syncthreads();
  }
  if (threadIdx.x == 0) {
    float m = sh[0] / n;
    mean[c] = m;
    var[c]  = sh2[0] / n - m * m;          // biased variance
  }
}

__global__ void k_bn_apply(float* __restrict__ x, const float* __restrict__ mean,
                           const float* __restrict__ var, const float* __restrict__ g,
                           const float* __restrict__ b, int n, int C) {
  long long tot = (long long)n * C;
  long long i = (long long)blockIdx.x * blockDim.x + threadIdx.x;
  long long stride = (long long)gridDim.x * blockDim.x;
  for (; i < tot; i += stride) {
    int c = (int)(i % C);
    float v = (x[i] - mean[c]) * rsqrtf(var[c] + BN_EPS) * g[c] + b[c];
    x[i] = v > 0.0f ? v : v * NEG_SLOPE;   // leaky relu
  }
}

// ---------------- last layer (512 -> 1): dot products + scalar aggregate ----------------
__global__ void k_dot2(const float* __restrict__ x, const float* __restrict__ wl,
                       const float* __restrict__ wr, float* __restrict__ s,
                       float* __restrict__ r, int n, int K) {
  int node = blockIdx.x * (blockDim.x >> 5) + (threadIdx.x >> 5);
  int lane = threadIdx.x & 31;
  if (node >= n) return;
  float a = 0.0f, b = 0.0f;
  for (int k = lane; k < K; k += 32) {
    float xv = x[(long long)node * K + k];
    a += xv * wl[k];
    b += xv * wr[k];
  }
  for (int o = 16; o > 0; o >>= 1) {
    a += __shfl_down(a, o, 32);
    b += __shfl_down(b, o, 32);
  }
  if (lane == 0) { s[node] = a; r[node] = b; }
}

__global__ void k_final(const float* __restrict__ t, const float* __restrict__ r,
                        const float* __restrict__ inv, const float* __restrict__ bl,
                        float* __restrict__ out, int n) {
  int i = blockIdx.x * blockDim.x + threadIdx.x;
  if (i < n) {
    float v = t[i] * inv[i] + r[i] + bl[0];
    out[i] = 1.0f / (1.0f + __expf(-v));
  }
}

// ---------------- host launch ----------------
extern "C" void kernel_launch(void* const* d_in, const int* in_sizes, int n_in,
                              void* d_out, int out_size, void* d_ws, size_t ws_size,
                              hipStream_t stream) {
  const float* x  = (const float*)d_in[0];
  const int* ei   = (const int*)d_in[1];
  const int* src  = ei;
  const int* dst  = ei + NE;
  const float *Wl[6], *bl[6], *Wr[6];
  for (int l = 0; l < 6; ++l) {
    Wl[l] = (const float*)d_in[2 + 3 * l];
    bl[l] = (const float*)d_in[3 + 3 * l];
    Wr[l] = (const float*)d_in[4 + 3 * l];
  }
  const float* g1 = (const float*)d_in[20];
  const float* b1 = (const float*)d_in[21];
  const float* g2 = (const float*)d_in[22];
  const float* b2 = (const float*)d_in[23];
  float* out = (float*)d_out;

  float* ws   = (float*)d_ws;
  float* inv  = ws;                                   // N
  float* agg  = ws + NN;                              // N*256 (also s/r/t for L5)
  float* P    = agg + (size_t)NN * 256;               // N*512
  float* Q    = P + (size_t)NN * 512;                 // N*512
  float* mean = Q + (size_t)NN * 512;                 // 512
  float* var  = mean + 512;                           // 512
  __bf16* Wt1 = (__bf16*)(var + 512);                 // 512*512 bf16
  __bf16* Wt2 = Wt1 + 512 * 512;                      // 512*512 bf16
  __bf16* Abf = Wt2 + 512 * 512;                      // N*256 bf16 (scaled aggregate)
  __bf16* Xbf = Abf + (size_t)NN * 256;               // N*256 bf16 (self features)

  auto zero = [&](float* p, long long n) {
    long long g = (n + 255) / 256; if (g > 8192) g = 8192;
    k_zero<<<(int)g, 256, 0, stream>>>(p, n);
  };
  auto sage = [&](const float* in, const float* Wl_, const float* bl_, const float* Wr_,
                  float* o, int din, int dout, int act) {
    zero(agg, (long long)NN * din);
    long long tot = (long long)NE * din;
    long long g = (tot + 255) / 256; if (g > 8192) g = 8192;
    k_scatter_add<<<(int)g, 256, 0, stream>>>(in, src, dst, agg, NE, din);
    long long nd = (long long)NN * din;
    long long gg = (nd + 255) / 256; if (gg > 8192) gg = 8192;
    k_agg_to_bf16<<<(int)gg, 256, 0, stream>>>(agg, inv, Abf, NN, din);
    k_to_bf16<<<(int)gg, 256, 0, stream>>>(in, Xbf, nd);
    int wtot = din * dout;
    k_w_transpose<<<(wtot + 255) / 256, 256, 0, stream>>>(Wl_, Wt1, din, dout);
    k_w_transpose<<<(wtot + 255) / 256, 256, 0, stream>>>(Wr_, Wt2, din, dout);
    dim3 grid((NN + 63) / 64, dout / 64);
    k_sage_gemm<<<grid, 128, 0, stream>>>(Abf, Xbf, Wt1, bl_, Wt2, o, din, dout, act);
  };

  // degree -> inverse counts
  zero(inv, NN);
  k_count<<<(NE + 255) / 256, 256, 0, stream>>>(dst, inv, NE);
  k_invert<<<(NN + 255) / 256, 256, 0, stream>>>(inv, NN);

  // gcn: SAGE(3,64)+relu -> SAGE(64,64) -> bn1 + leaky
  zero(agg, (long long)NN * 3);
  k_scatter_add<<<(NE * 3 + 255) / 256, 256, 0, stream>>>(x, src, dst, agg, NE, 3);
  k_sage3<<<(NN * 64 + 255) / 256, 256, 0, stream>>>(agg, x, inv, Wl[0], bl[0], Wr[0], P, NN);

  sage(P, Wl[1], bl[1], Wr[1], Q, 64, 64, 0);
  k_bn_stats<<<64, 256, 0, stream>>>(Q, mean, var, NN, 64);
  k_bn_apply<<<8192, 256, 0, stream>>>(Q, mean, var, g1, b1, NN, 64);

  // gcn1: SAGE(64,256)+relu -> SAGE(256,256) -> bn2 + leaky
  sage(Q, Wl[2], bl[2], Wr[2], P, 64, 256, 1);
  sage(P, Wl[3], bl[3], Wr[3], Q, 256, 256, 0);
  k_bn_stats<<<256, 256, 0, stream>>>(Q, mean, var, NN, 256);
  k_bn_apply<<<8192, 256, 0, stream>>>(Q, mean, var, g2, b2, NN, 256);

  // gcn2: SAGE(256,512)+relu
  sage(Q, Wl[4], bl[4], Wr[4], P, 256, 512, 1);

  // final SAGE(512,1): aggregate AFTER the linear map (mean commutes with Wl)
  float* s5 = agg;
  float* r5 = agg + NN;
  float* t5 = agg + 2 * NN;
  k_dot2<<<(NN + 7) / 8, 256, 0, stream>>>(P, Wl[5], Wr[5], s5, r5, NN, 512);
  zero(t5, NN);
  k_scatter_add<<<(NE + 255) / 256, 256, 0, stream>>>(s5, src, dst, t5, NE, 1);
  k_final<<<(NN + 255) / 256, 256, 0, stream>>>(t5, r5, inv, bl[5], out, NN);
}
